// GCNModel_63196148793943
// MI455X (gfx1250) — matrix operand order repaired and checked by reading, not compiled, not run on hardware
//
#include <hip/hip_runtime.h>
#include <hip/hip_bf16.h>

typedef __attribute__((ext_vector_type(16))) __bf16 v16bf;
typedef __attribute__((ext_vector_type(8)))  float  v8f;

#define N_NODES 100000
#define N_EDGES 1600000
#define D       128
#define N_GRAPH 512
#define ND      (N_NODES * D)
#define BN_EPS  1e-5f
#define WPACK_ELEMS (8 * 4 * 32 * 16)   // tiles * kb * lanes * elems = 16384

// ---------------------------------------------------------------- utilities
__global__ void fill_zero_kernel(float* __restrict__ p, int n) {
    int i = blockIdx.x * blockDim.x + threadIdx.x;
    if (i < n) p[i] = 0.0f;
}

// deg[dst] += 1 per edge
__global__ void deg_accum_kernel(const int* __restrict__ dst, float* __restrict__ deg) {
    int e = blockIdx.x * blockDim.x + threadIdx.x;
    if (e < N_EDGES) atomicAdd(&deg[dst[e]], 1.0f);
}

// in-place: deg -> rsqrt(deg + 2)
__global__ void dinv_kernel(float* __restrict__ deg) {
    int i = blockIdx.x * blockDim.x + threadIdx.x;
    if (i < N_NODES) deg[i] = rsqrtf(deg[i] + 2.0f);
}

// ---------------------------------------------------------------- W pre-pack
// Repack W[128x128] (K x N, row-major fp32) into bf16 WMMA B-fragment order:
// P[((n_tile*4 + kb)*32 + lane)*16 + j]
//   = bf16( W[(kb*32 + 16*(lane>=16) + j)*128 + n_tile*16 + (lane&15)] )
// so each lane's fragment is 16 contiguous bf16 (32B) -> one vector load.
__global__ void pack_w_kernel(const float* __restrict__ W, __bf16* __restrict__ P) {
    const int idx = blockIdx.x * blockDim.x + threadIdx.x;
    if (idx >= WPACK_ELEMS) return;
    const int j    = idx & 15;
    const int lane = (idx >> 4) & 31;
    const int kb   = (idx >> 9) & 3;
    const int n    = idx >> 11;
    const int khalf = (lane >> 4) << 4;
    const int col   = n * 16 + (lane & 15);
    P[idx] = (__bf16)W[(kb * 32 + khalf + j) * 128 + col];
}

// ---------------------------------------------------------------- GEMM (WMMA)
// H[N_NODES x 128] = X[N_NODES x 128] @ W[128 x 128]  (W pre-packed bf16)
// block = 128 threads (4 waves). Block owns a 16-row stripe of X staged in LDS.
// Wave w computes column tiles 2w and 2w+1 (cols [32w, 32w+32)).
__global__ void gemm_wmma_kernel(const float* __restrict__ X,
                                 const __bf16* __restrict__ Wp,
                                 float* __restrict__ H) {
    __shared__ float tile[16 * 128]; // 8KB: 16 rows of X
    const int tid = threadIdx.x;
    const int r0  = blockIdx.x * 16;

    // coalesced cooperative load: 16 rows x 128 cols
    #pragma unroll
    for (int i = 0; i < 16; ++i)
        tile[i * 128 + tid] = X[(r0 + i) * 128 + tid];
    __syncthreads();

    const int wave = tid >> 5;
    const int lane = tid & 31;
    const int m    = lane & 15;     // A row within tile / C col
    const int hhi  = lane >> 4;     // lane-half select
    const int n0   = wave * 2;      // B column tiles
    const int n1   = wave * 2 + 1;
    const v16bf* P = (const v16bf*)Wp;

    v8f c0 = {0.f,0.f,0.f,0.f,0.f,0.f,0.f,0.f};
    v8f c1 = {0.f,0.f,0.f,0.f,0.f,0.f,0.f,0.f};

    #pragma unroll
    for (int kb = 0; kb < 4; ++kb) {
        const int kb32 = kb * 32;

        // A fragment: contiguous k order, same as packed B
        // lanes 0-15: K = 0..15; lanes 16-31: K = 16..31
        v16bf a;
        const int ah = hhi * 8;
        #pragma unroll
        for (int j = 0; j < 16; ++j) {
            const int g = j >> 1, p = j & 1;
            const int kl = ah * 2 + g * 2 + p;
            a[j] = (__bf16)tile[m * 128 + kb32 + kl];
        }

        // B fragments: one 32-byte vector load each (pre-packed)
        const v16bf b0 = P[(n0 * 4 + kb) * 32 + lane];
        const v16bf b1 = P[(n1 * 4 + kb) * 32 + lane];

        c0 = __builtin_amdgcn_wmma_f32_16x16x32_bf16(false, a, false, b0, (short)0, c0, false, false);
        c1 = __builtin_amdgcn_wmma_f32_16x16x32_bf16(false, a, false, b1, (short)0, c1, false, false);
    }

    // C/D layout: VGPR r -> M = r + 8*(lane>=16), N = lane&15
    const int mrow = hhi << 3;
    #pragma unroll
    for (int r = 0; r < 8; ++r) {
        const int row = r0 + r + mrow;
        H[row * 128 + n0 * 16 + m] = c0[r];
        H[row * 128 + n1 * 16 + m] = c1[r];
    }
}

// ---------------------------------------------------------------- edge scatter
// AGG[dst] += H[src] * (dinv[src]*dinv[dst]); one wave per edge, float4 per lane
__global__ void edge_scatter_kernel(const int* __restrict__ src,
                                    const int* __restrict__ dst,
                                    const float* __restrict__ dinv,
                                    const float* __restrict__ H,
                                    float* __restrict__ AGG) {
    const int wave = threadIdx.x >> 5;
    const int lane = threadIdx.x & 31;
    const int e = blockIdx.x * 8 + wave;
    if (e >= N_EDGES) return;
    const int s = src[e];
    const int d = dst[e];
    const float norm = dinv[s] * dinv[d];
    const float4 v = *(const float4*)(H + s * 128 + lane * 4);
    float* out = AGG + d * 128 + lane * 4;
    atomicAdd(out + 0, v.x * norm);
    atomicAdd(out + 1, v.y * norm);
    atomicAdd(out + 2, v.z * norm);
    atomicAdd(out + 3, v.w * norm);
}

// OUT = AGG + H * (2*dinv^2) + bias   (float4 per thread)
__global__ void finalize_conv_kernel(const float* __restrict__ AGG,
                                     const float* __restrict__ H,
                                     const float* __restrict__ dinv,
                                     const float* __restrict__ bias,
                                     float* __restrict__ OUT) {
    const int idx = blockIdx.x * blockDim.x + threadIdx.x; // over ND/4
    if (idx >= ND / 4) return;
    const int i = idx * 4;
    const int node = i >> 7;
    const int c = i & 127;
    const float di = dinv[node];
    const float coef = 2.0f * di * di;
    const float4 a = *(const float4*)(AGG + i);
    const float4 h = *(const float4*)(H + i);
    float4 o;
    o.x = a.x + h.x * coef + bias[c + 0];
    o.y = a.y + h.y * coef + bias[c + 1];
    o.z = a.z + h.z * coef + bias[c + 2];
    o.w = a.w + h.w * coef + bias[c + 3];
    *(float4*)(OUT + i) = o;
}

// ---------------------------------------------------------------- batch norm
// block = 256 threads = 2 rows x 128 cols; block covers 128 rows
__global__ void bn_stats_kernel(const float* __restrict__ X,
                                float* __restrict__ colsum,
                                float* __restrict__ colsq) {
    __shared__ float ss[256];
    __shared__ float sq[256];
    const int t = threadIdx.x;
    const int col = t & 127;
    const int half = t >> 7;
    float s = 0.f, q = 0.f;
    #pragma unroll 4
    for (int i = 0; i < 64; ++i) {
        const int row = blockIdx.x * 128 + half + i * 2;
        if (row < N_NODES) {
            const float v = X[row * 128 + col];
            s += v;
            q += v * v;
        }
    }
    ss[t] = s; sq[t] = q;
    __syncthreads();
    if (t < 128) {
        atomicAdd(&colsum[t], ss[t] + ss[t + 128]);
        atomicAdd(&colsq[t],  sq[t] + sq[t + 128]);
    }
}

__global__ void bn_coeff_kernel(const float* __restrict__ colsum,
                                const float* __restrict__ colsq,
                                const float* __restrict__ gamma,
                                const float* __restrict__ beta,
                                float* __restrict__ scale,
                                float* __restrict__ shift) {
    const int c = threadIdx.x;
    if (c < 128) {
        const float inv_n = 1.0f / (float)N_NODES;
        const float mean = colsum[c] * inv_n;
        const float var = colsq[c] * inv_n - mean * mean;
        const float sc = gamma[c] * rsqrtf(var + BN_EPS);
        scale[c] = sc;
        shift[c] = beta[c] - mean * sc;
    }
}

// OUT = relu(X*scale + shift) + RES   (float4 per thread)
__global__ void bn_apply_relu_res_kernel(const float* __restrict__ X,
                                         const float* __restrict__ scale,
                                         const float* __restrict__ shift,
                                         const float* __restrict__ RES,
                                         float* __restrict__ OUT) {
    const int idx = blockIdx.x * blockDim.x + threadIdx.x;
    if (idx >= ND / 4) return;
    const int i = idx * 4;
    const int c = i & 127;
    const float4 x = *(const float4*)(X + i);
    const float4 r = *(const float4*)(RES + i);
    float4 o;
    o.x = fmaxf(x.x * scale[c + 0] + shift[c + 0], 0.f) + r.x;
    o.y = fmaxf(x.y * scale[c + 1] + shift[c + 1], 0.f) + r.y;
    o.z = fmaxf(x.z * scale[c + 2] + shift[c + 2], 0.f) + r.z;
    o.w = fmaxf(x.w * scale[c + 3] + shift[c + 3], 0.f) + r.w;
    *(float4*)(OUT + i) = o;
}

// ---------------------------------------------------------------- pooling + fc
__global__ void pool_kernel(const float* __restrict__ H,
                            const int* __restrict__ batch,
                            float* __restrict__ pooled) {
    const int idx = blockIdx.x * blockDim.x + threadIdx.x;
    if (idx >= ND) return;
    const int node = idx >> 7;
    const int c = idx & 127;
    const int g = batch[node];
    atomicAdd(&pooled[g * 128 + c], H[idx]);
}

__global__ void fc_kernel(const float* __restrict__ pooled,
                          const float* __restrict__ fcW,
                          const float* __restrict__ fcb,
                          float* __restrict__ out) {
    const int wave = threadIdx.x >> 5;
    const int lane = threadIdx.x & 31;
    const int g = blockIdx.x * 8 + wave;
    if (g >= N_GRAPH) return;
    float acc = 0.f;
    #pragma unroll
    for (int j = 0; j < 4; ++j) {
        const int c = lane + 32 * j;
        acc += pooled[g * 128 + c] * fcW[c];
    }
    #pragma unroll
    for (int off = 16; off > 0; off >>= 1)
        acc += __shfl_xor(acc, off, 32);
    if (lane == 0) out[g] = acc + fcb[0];
}

// ---------------------------------------------------------------- launch
extern "C" void kernel_launch(void* const* d_in, const int* in_sizes, int n_in,
                              void* d_out, int out_size, void* d_ws, size_t ws_size,
                              hipStream_t stream) {
    const float* x   = (const float*)d_in[0];
    const int*   ei  = (const int*)d_in[1];   // [2, E] flat
    const int*   bat = (const int*)d_in[2];
    const float* W0  = (const float*)d_in[3];
    const float* b0  = (const float*)d_in[4];
    const float* g0  = (const float*)d_in[5];
    const float* be0 = (const float*)d_in[6];
    const float* W1  = (const float*)d_in[7];
    const float* b1  = (const float*)d_in[8];
    const float* g1  = (const float*)d_in[9];
    const float* be1 = (const float*)d_in[10];
    const float* W2  = (const float*)d_in[11];
    const float* b2  = (const float*)d_in[12];
    const float* fcW = (const float*)d_in[13];
    const float* fcb = (const float*)d_in[14];
    float* out = (float*)d_out;

    const int* src = ei;
    const int* dst = ei + N_EDGES;

    // workspace layout (floats; every offset is a multiple of 128 -> 512B aligned)
    float* ws     = (float*)d_ws;
    float* dinv   = ws;                     // 100352 (rounded up)
    float* colsum = dinv + 100352;          // 128
    float* colsq  = colsum + 128;           // 128
    float* scale  = colsq + 128;            // 128
    float* shift  = scale + 128;            // 128
    float* pooled = shift + 128;            // 65536
    __bf16* Wp0   = (__bf16*)(pooled + 65536);          // 16384 bf16 = 8192 floats
    __bf16* Wp1   = Wp0 + WPACK_ELEMS;
    __bf16* Wp2   = Wp1 + WPACK_ELEMS;
    float* buf1   = pooled + 65536 + 3 * (WPACK_ELEMS / 2); // ND
    float* buf2   = buf1 + ND;                              // ND
    float* buf3   = buf2 + ND;                              // ND

    const dim3 blk256(256);
    const int gN    = (N_NODES + 255) / 256;
    const int gE    = (N_EDGES + 255) / 256;
    const int gND4  = (ND / 4 + 255) / 256;
    const int gND   = (ND + 255) / 256;
    const int gEdge = (N_EDGES + 7) / 8;       // 8 waves per block of 256
    const int gBN   = (N_NODES + 127) / 128;
    const int gGemm = N_NODES / 16;            // 6250 (exact)
    const int gPack = (WPACK_ELEMS + 255) / 256;

    // degree -> dinv; pre-pack all three weight matrices to bf16 fragment order
    fill_zero_kernel<<<gN, blk256, 0, stream>>>(dinv, N_NODES);
    deg_accum_kernel<<<gE, blk256, 0, stream>>>(dst, dinv);
    dinv_kernel<<<gN, blk256, 0, stream>>>(dinv);
    pack_w_kernel<<<gPack, blk256, 0, stream>>>(W0, Wp0);
    pack_w_kernel<<<gPack, blk256, 0, stream>>>(W1, Wp1);
    pack_w_kernel<<<gPack, blk256, 0, stream>>>(W2, Wp2);

    // ---------------- layer 1 ----------------
    gemm_wmma_kernel<<<gGemm, 128, 0, stream>>>(x, Wp0, buf1);
    fill_zero_kernel<<<gND, blk256, 0, stream>>>(buf2, ND);
    edge_scatter_kernel<<<gEdge, blk256, 0, stream>>>(src, dst, dinv, buf1, buf2);
    finalize_conv_kernel<<<gND4, blk256, 0, stream>>>(buf2, buf1, dinv, b0, buf2);
    fill_zero_kernel<<<1, blk256, 0, stream>>>(colsum, 256);   // colsum+colsq
    bn_stats_kernel<<<gBN, blk256, 0, stream>>>(buf2, colsum, colsq);
    bn_coeff_kernel<<<1, 128, 0, stream>>>(colsum, colsq, g0, be0, scale, shift);
    bn_apply_relu_res_kernel<<<gND4, blk256, 0, stream>>>(buf2, scale, shift, x, buf3);

    // ---------------- layer 2 ----------------
    gemm_wmma_kernel<<<gGemm, 128, 0, stream>>>(buf3, Wp1, buf1);
    fill_zero_kernel<<<gND, blk256, 0, stream>>>(buf2, ND);
    edge_scatter_kernel<<<gEdge, blk256, 0, stream>>>(src, dst, dinv, buf1, buf2);
    finalize_conv_kernel<<<gND4, blk256, 0, stream>>>(buf2, buf1, dinv, b1, buf2);
    fill_zero_kernel<<<1, blk256, 0, stream>>>(colsum, 256);
    bn_stats_kernel<<<gBN, blk256, 0, stream>>>(buf2, colsum, colsq);
    bn_coeff_kernel<<<1, 128, 0, stream>>>(colsum, colsq, g1, be1, scale, shift);
    bn_apply_relu_res_kernel<<<gND4, blk256, 0, stream>>>(buf2, scale, shift, buf3, buf1);

    // ---------------- layer 3 ----------------
    gemm_wmma_kernel<<<gGemm, 128, 0, stream>>>(buf1, Wp2, buf3);
    fill_zero_kernel<<<gND, blk256, 0, stream>>>(buf2, ND);
    edge_scatter_kernel<<<gEdge, blk256, 0, stream>>>(src, dst, dinv, buf3, buf2);
    finalize_conv_kernel<<<gND4, blk256, 0, stream>>>(buf2, buf3, dinv, b2, buf2);

    // pool + fc
    fill_zero_kernel<<<(N_GRAPH * 128 + 255) / 256, blk256, 0, stream>>>(pooled, N_GRAPH * 128);
    pool_kernel<<<gND, blk256, 0, stream>>>(buf2, bat, pooled);
    fc_kernel<<<N_GRAPH / 8, blk256, 0, stream>>>(pooled, fcW, fcb, out);
}
